// DAWN_37701222924826
// MI455X (gfx1250) — compile-verified, matrix operand order checked by reading, not run on hardware
//
#include <hip/hip_runtime.h>
#include <hip/hip_bf16.h>
#include <math.h>

typedef __bf16 bf16_t;
typedef __attribute__((ext_vector_type(16))) __bf16 v16bf;
typedef __attribute__((ext_vector_type(8)))  __bf16 v8bf;
typedef __attribute__((ext_vector_type(8)))  float  v8f;

#define M_TOT 4096
#define D_DIM 1024
#define N_TOT 32768

#define BLK_M  64
#define BLK_N  128
#define BLK_ND 256      // output-GEMM D tile
#define KSTEP  32
#define LDA    40       // LDS row stride (bf16), 80B: 16B-aligned + bank skew

// d_out layout (f32, reference return order)
#define OFF_OUT 0
#define OFF_ACT 4194304
#define OFF_GMX 4198400
#define OFF_SC  4202496  // +0 score_lb, +1 score_std, +2 es, +3 active_n_mean, +4 score_mean

enum : unsigned {
  A_BYTES  = BLK_M  * LDA * 2,   // 5120
  B_BYTES  = BLK_N  * LDA * 2,   // 10240
  BD_BYTES = BLK_ND * LDA * 2,   // 20480
  CS_BYTES = BLK_M * (BLK_N + 4) * 4,                    // 33792
  SM_STATS = (2 * A_BYTES + 2 * B_BYTES > CS_BYTES) ? (2 * A_BYTES + 2 * B_BYTES) : CS_BYTES,
  SM_GATE  = 4 * A_BYTES + 4 * B_BYTES,                  // 61440 (> CS_BYTES)
  SM_OUT   = 2 * A_BYTES + 2 * BD_BYTES                  // 51200
};

union FragU { v16bf v; v8bf h[2]; };

// 16x32 bf16 fragment from LDS (K-contiguous rows, stride LDA).
// ISA layout: lane<16 -> row=lane, K 0..7 & 16..23; lane>=16 -> K 8..15 & 24..31.
__device__ __forceinline__ v16bf frag_ld(const bf16_t* tile, int lane) {
  const int row = lane & 15;
  const int kb  = (lane >> 4) << 3;
  FragU f;
  f.h[0] = *(const v8bf*)(tile + row * LDA + kb);
  f.h[1] = *(const v8bf*)(tile + row * LDA + 16 + kb);
  return f.v;
}

__device__ __forceinline__ v8f wmma_bf16(v16bf a, v16bf b, v8f c) {
  return __builtin_amdgcn_wmma_f32_16x16x32_bf16(false, a, false, b, (short)0, c, false, false);
}

// ---------------------------------------------------------------- prep: f32 -> bf16 (+row norm)
__global__ __launch_bounds__(256) void k_rownorm_bf16(const float* __restrict__ in,
                                                      bf16_t* __restrict__ out, int do_norm) {
  const int row = blockIdx.x;
  const int tid = threadIdx.x;
  const float* rp = in + (size_t)row * D_DIM;
  __shared__ float red[256];
  float scale = 1.f;
  if (do_norm) {
    float ss = 0.f;
    for (int c = tid; c < D_DIM; c += 256) { float v = rp[c]; ss += v * v; }
    red[tid] = ss; __syncthreads();
    for (int s = 128; s > 0; s >>= 1) { if (tid < s) red[tid] += red[tid + s]; __syncthreads(); }
    scale = 1.f / (sqrtf(red[0]) + 1e-8f);
  }
  for (int c = tid; c < D_DIM; c += 256)
    out[(size_t)row * D_DIM + c] = (bf16_t)(rp[c] * scale);
}

// ---------------------------------------------------------------- 64x64 LDS-tiled transpose [N][D]->[D][N]
__global__ __launch_bounds__(256) void k_transpose(const bf16_t* __restrict__ in,
                                                   bf16_t* __restrict__ out) {
  __shared__ bf16_t tile[64][72];
  const int n0 = blockIdx.x * 64, d0 = blockIdx.y * 64;
  const int tid = threadIdx.x;
  const int r = tid >> 2, c = (tid & 3) * 16;
  v8bf u0 = *(const v8bf*)(in + (size_t)(n0 + r) * D_DIM + d0 + c);
  v8bf u1 = *(const v8bf*)(in + (size_t)(n0 + r) * D_DIM + d0 + c + 8);
  *(v8bf*)(&tile[r][c])     = u0;
  *(v8bf*)(&tile[r][c + 8]) = u1;
  __syncthreads();
  v8bf o0, o1;
#pragma unroll
  for (int j = 0; j < 8; ++j) { o0[j] = tile[c + j][r]; o1[j] = tile[c + 8 + j][r]; }
  *(v8bf*)(out + (size_t)(d0 + r) * N_TOT + n0 + c)     = o0;
  *(v8bf*)(out + (size_t)(d0 + r) * N_TOT + n0 + c + 8) = o1;
}

// ---------------------------------------------------------------- pass 1: scores stats GEMM (double-buffered)
__global__ __launch_bounds__(256) void k_stats_gemm(const bf16_t* __restrict__ hb,
                                                    const bf16_t* __restrict__ eb,
                                                    float* __restrict__ s_sum,
                                                    float* __restrict__ sq_sum,
                                                    float* __restrict__ colsum) {
  __shared__ __align__(16) char smraw[SM_STATS];

  const int m0 = blockIdx.x * BLK_M, n0 = blockIdx.y * BLK_N;
  const int tid = threadIdx.x, wave = tid >> 5, lane = tid & 31;
  const int wm = (wave & 1) * 32, wn = (wave >> 1) * 32;
  const int ar = tid >> 2, ac = (tid & 3) * 8;     // A: 64x32
  const int br = tid >> 1, bc = (tid & 1) * 16;    // B: 128x32
  const bf16_t* ap = hb + (size_t)(m0 + ar) * D_DIM + ac;
  const bf16_t* bp = eb + (size_t)(n0 + br) * D_DIM + bc;

  { // prologue: tile 0 -> buffer 0
    bf16_t* A0 = (bf16_t*)smraw;
    bf16_t* B0 = (bf16_t*)(smraw + 2 * A_BYTES);
    v8bf av = *(const v8bf*)(ap);
    v8bf b0 = *(const v8bf*)(bp);
    v8bf b1 = *(const v8bf*)(bp + 8);
    *(v8bf*)(A0 + ar * LDA + ac) = av;
    *(v8bf*)(B0 + br * LDA + bc) = b0;
    *(v8bf*)(B0 + br * LDA + bc + 8) = b1;
  }
  __syncthreads();

  v8f acc[2][2] = {};
  for (int k0 = 0; k0 < D_DIM; k0 += KSTEP) {
    const int cur = (k0 / KSTEP) & 1, nxt = cur ^ 1;
    const bool more = (k0 + KSTEP) < D_DIM;
    bf16_t* Acur = (bf16_t*)(smraw + cur * A_BYTES);
    bf16_t* Anxt = (bf16_t*)(smraw + nxt * A_BYTES);
    bf16_t* Bcur = (bf16_t*)(smraw + 2 * A_BYTES + cur * B_BYTES);
    bf16_t* Bnxt = (bf16_t*)(smraw + 2 * A_BYTES + nxt * B_BYTES);
    v8bf av = {}, b0 = {}, b1 = {};
    if (more) {
      av = *(const v8bf*)(ap + k0 + KSTEP);
      b0 = *(const v8bf*)(bp + k0 + KSTEP);
      b1 = *(const v8bf*)(bp + k0 + KSTEP + 8);
    }
    v16bf a0 = frag_ld(Acur + (wm + 0)  * LDA, lane);
    v16bf a1 = frag_ld(Acur + (wm + 16) * LDA, lane);
    v16bf e0 = frag_ld(Bcur + (wn + 0)  * LDA, lane);
    v16bf e1 = frag_ld(Bcur + (wn + 16) * LDA, lane);
    acc[0][0] = wmma_bf16(a0, e0, acc[0][0]);
    acc[0][1] = wmma_bf16(a0, e1, acc[0][1]);
    acc[1][0] = wmma_bf16(a1, e0, acc[1][0]);
    acc[1][1] = wmma_bf16(a1, e1, acc[1][1]);
    if (more) {
      *(v8bf*)(Anxt + ar * LDA + ac) = av;
      *(v8bf*)(Bnxt + br * LDA + bc) = b0;
      *(v8bf*)(Bnxt + br * LDA + bc + 8) = b1;
    }
    __syncthreads();
  }

  float (*Cs)[BLK_N + 4] = (float (*)[BLK_N + 4])smraw;   // aliases tiles, live after K-loop
  const int crow = (lane >> 4) * 8, ccol = lane & 15;
#pragma unroll
  for (int ti = 0; ti < 2; ++ti)
#pragma unroll
    for (int tj = 0; tj < 2; ++tj)
#pragma unroll
      for (int i = 0; i < 8; ++i)
        Cs[wm + ti * 16 + crow + i][wn + tj * 16 + ccol] = acc[ti][tj][i];
  __syncthreads();
  if (tid < 64) {
    float rs = 0.f, rq = 0.f;
    for (int c = 0; c < BLK_N; ++c) { float v = Cs[tid][c]; rs += v; rq += v * v; }
    atomicAdd(&s_sum[m0 + tid], rs);
    atomicAdd(&sq_sum[m0 + tid], rq);
  } else if (tid < 192) {
    const int c = tid - 64;
    float cs = 0.f;
    for (int r = 0; r < BLK_M; ++r) cs += Cs[r][c];
    atomicAdd(&colsum[n0 + c], cs);
  }
}

// ---------------------------------------------------------------- stats finalize
__global__ __launch_bounds__(256) void k_finalize_stats(const float* __restrict__ s_sum,
                                                        const float* __restrict__ sq_sum,
                                                        const float* __restrict__ colsum,
                                                        const float* __restrict__ tau_off,
                                                        float* __restrict__ tau,
                                                        float* __restrict__ sstd,
                                                        float* __restrict__ outp) {
  __shared__ float r1[256], r2[256];
  const int tid = threadIdx.x;
  float macc = 0.f, sacc = 0.f;
  for (int m = tid; m < M_TOT; m += 256) {
    float mean = s_sum[m] * (1.f / N_TOT);
    float var  = sq_sum[m] * (1.f / N_TOT) - mean * mean;
    float sd   = sqrtf(fmaxf(var, 0.f)) + 1e-8f;
    tau[m]  = mean + tau_off[m] * sd;
    sstd[m] = sd;
    macc += mean; sacc += sd;
  }
  r1[tid] = macc; r2[tid] = sacc; __syncthreads();
  for (int s = 128; s > 0; s >>= 1) {
    if (tid < s) { r1[tid] += r1[tid + s]; r2[tid] += r2[tid + s]; }
    __syncthreads();
  }
  if (tid == 0) {
    outp[OFF_SC + 4] = r1[0] * (1.f / M_TOT);
    outp[OFF_SC + 1] = r2[0] * (1.f / M_TOT);
  }
  __syncthreads();
  float ns = 0.f, nq = 0.f;
  for (int n = tid; n < N_TOT; n += 256) {
    float p = colsum[n] * (1.f / M_TOT);
    ns += p; nq += p * p;
  }
  r1[tid] = ns; r2[tid] = nq; __syncthreads();
  for (int s = 128; s > 0; s >>= 1) {
    if (tid < s) { r1[tid] += r1[tid + s]; r2[tid] += r2[tid + s]; }
    __syncthreads();
  }
  if (tid == 0) {
    float ms = r1[0] * (1.f / N_TOT);
    float vs = r2[0] * (1.f / N_TOT) - ms * ms;
    outp[OFF_SC + 0] = vs / (ms * ms + vs + 0.01f);
  }
}

// ---------------------------------------------------------------- pass 2: fused score+xr, gating (double-buffered)
__global__ __launch_bounds__(256) void k_gate_gemm(const bf16_t* __restrict__ hb,
                                                   const bf16_t* __restrict__ xb,
                                                   const bf16_t* __restrict__ eb,
                                                   const bf16_t* __restrict__ rb,
                                                   const float* __restrict__ tau,
                                                   const float* __restrict__ sstd,
                                                   bf16_t* __restrict__ tbuf,
                                                   float* __restrict__ twc,
                                                   float* __restrict__ tact) {
  __shared__ __align__(16) char smraw[SM_GATE];
  __shared__ float tau_s[BLK_M], std_s[BLK_M];

  const int m0 = blockIdx.x * BLK_M, n0 = blockIdx.y * BLK_N;
  const int tid = threadIdx.x, wave = tid >> 5, lane = tid & 31;
  const int wm = (wave & 1) * 32, wn = (wave >> 1) * 32;
  const int ar = tid >> 2, ac = (tid & 3) * 8;
  const int br = tid >> 1, bc = (tid & 1) * 16;
  const bf16_t* hp = hb + (size_t)(m0 + ar) * D_DIM + ac;
  const bf16_t* xp = xb + (size_t)(m0 + ar) * D_DIM + ac;
  const bf16_t* ep = eb + (size_t)(n0 + br) * D_DIM + bc;
  const bf16_t* rp = rb + (size_t)(n0 + br) * D_DIM + bc;
  if (tid < BLK_M) { tau_s[tid] = tau[m0 + tid]; std_s[tid] = sstd[m0 + tid]; }

  { // prologue: buffers 0 (layout: Ah[2] | Ax[2] | Be[2] | Br[2])
    bf16_t* Ah0 = (bf16_t*)smraw;
    bf16_t* Ax0 = (bf16_t*)(smraw + 2 * A_BYTES);
    bf16_t* Be0 = (bf16_t*)(smraw + 4 * A_BYTES);
    bf16_t* Br0 = (bf16_t*)(smraw + 4 * A_BYTES + 2 * B_BYTES);
    v8bf h0 = *(const v8bf*)(hp);
    v8bf x0 = *(const v8bf*)(xp);
    v8bf e0 = *(const v8bf*)(ep), e1 = *(const v8bf*)(ep + 8);
    v8bf r0 = *(const v8bf*)(rp), r1 = *(const v8bf*)(rp + 8);
    *(v8bf*)(Ah0 + ar * LDA + ac) = h0;
    *(v8bf*)(Ax0 + ar * LDA + ac) = x0;
    *(v8bf*)(Be0 + br * LDA + bc) = e0;  *(v8bf*)(Be0 + br * LDA + bc + 8) = e1;
    *(v8bf*)(Br0 + br * LDA + bc) = r0;  *(v8bf*)(Br0 + br * LDA + bc + 8) = r1;
  }
  __syncthreads();

  v8f accs[2][2] = {}, accx[2][2] = {};
  for (int k0 = 0; k0 < D_DIM; k0 += KSTEP) {
    const int cur = (k0 / KSTEP) & 1, nxt = cur ^ 1;
    const bool more = (k0 + KSTEP) < D_DIM;
    bf16_t* AhC = (bf16_t*)(smraw + cur * A_BYTES);
    bf16_t* AhN = (bf16_t*)(smraw + nxt * A_BYTES);
    bf16_t* AxC = (bf16_t*)(smraw + 2 * A_BYTES + cur * A_BYTES);
    bf16_t* AxN = (bf16_t*)(smraw + 2 * A_BYTES + nxt * A_BYTES);
    bf16_t* BeC = (bf16_t*)(smraw + 4 * A_BYTES + cur * B_BYTES);
    bf16_t* BeN = (bf16_t*)(smraw + 4 * A_BYTES + nxt * B_BYTES);
    bf16_t* BrC = (bf16_t*)(smraw + 4 * A_BYTES + 2 * B_BYTES + cur * B_BYTES);
    bf16_t* BrN = (bf16_t*)(smraw + 4 * A_BYTES + 2 * B_BYTES + nxt * B_BYTES);
    v8bf nh = {}, nx = {}, ne0 = {}, ne1 = {}, nr0 = {}, nr1 = {};
    if (more) {
      nh  = *(const v8bf*)(hp + k0 + KSTEP);
      nx  = *(const v8bf*)(xp + k0 + KSTEP);
      ne0 = *(const v8bf*)(ep + k0 + KSTEP);  ne1 = *(const v8bf*)(ep + k0 + KSTEP + 8);
      nr0 = *(const v8bf*)(rp + k0 + KSTEP);  nr1 = *(const v8bf*)(rp + k0 + KSTEP + 8);
    }
    v16bf fh0 = frag_ld(AhC + (wm + 0)  * LDA, lane);
    v16bf fh1 = frag_ld(AhC + (wm + 16) * LDA, lane);
    v16bf fe0 = frag_ld(BeC + (wn + 0)  * LDA, lane);
    v16bf fe1 = frag_ld(BeC + (wn + 16) * LDA, lane);
    accs[0][0] = wmma_bf16(fh0, fe0, accs[0][0]);
    accs[0][1] = wmma_bf16(fh0, fe1, accs[0][1]);
    accs[1][0] = wmma_bf16(fh1, fe0, accs[1][0]);
    accs[1][1] = wmma_bf16(fh1, fe1, accs[1][1]);
    v16bf fx0 = frag_ld(AxC + (wm + 0)  * LDA, lane);
    v16bf fx1 = frag_ld(AxC + (wm + 16) * LDA, lane);
    v16bf fr0 = frag_ld(BrC + (wn + 0)  * LDA, lane);
    v16bf fr1 = frag_ld(BrC + (wn + 16) * LDA, lane);
    accx[0][0] = wmma_bf16(fx0, fr0, accx[0][0]);
    accx[0][1] = wmma_bf16(fx0, fr1, accx[0][1]);
    accx[1][0] = wmma_bf16(fx1, fr0, accx[1][0]);
    accx[1][1] = wmma_bf16(fx1, fr1, accx[1][1]);
    if (more) {
      *(v8bf*)(AhN + ar * LDA + ac) = nh;
      *(v8bf*)(AxN + ar * LDA + ac) = nx;
      *(v8bf*)(BeN + br * LDA + bc) = ne0;  *(v8bf*)(BeN + br * LDA + bc + 8) = ne1;
      *(v8bf*)(BrN + br * LDA + bc) = nr0;  *(v8bf*)(BrN + br * LDA + bc + 8) = nr1;
    }
    __syncthreads();
  }

  float (*Cs)[BLK_N + 4] = (float (*)[BLK_N + 4])smraw;   // aliases tiles, live after K-loop
  const int crow = (lane >> 4) * 8, ccol = lane & 15;
#pragma unroll
  for (int ti = 0; ti < 2; ++ti)
#pragma unroll
    for (int tj = 0; tj < 2; ++tj)
#pragma unroll
      for (int i = 0; i < 8; ++i) {
        const int rl = wm + ti * 16 + crow + i;
        const int cl = wn + tj * 16 + ccol;
        float s   = accs[ti][tj][i];
        float xr  = accx[ti][tj][i];
        float raw = s - tau_s[rl];
        float gh  = raw > 0.f ? 1.f : 0.f;
        float sig = 1.f / (1.f + __expf(-raw / std_s[rl]));
        tbuf[(size_t)(m0 + rl) * N_TOT + (n0 + cl)] = (bf16_t)(gh * xr);
        Cs[rl][cl] = sig * xr * xr;
      }
  __syncthreads();
  if (tid < 64) {
    float ws = 0.f;
    for (int c = 0; c < BLK_N; ++c) ws += Cs[tid][c];
    atomicAdd(&twc[m0 + tid], ws);
  }
  __syncthreads();
#pragma unroll
  for (int ti = 0; ti < 2; ++ti)
#pragma unroll
    for (int tj = 0; tj < 2; ++tj)
#pragma unroll
      for (int i = 0; i < 8; ++i) {
        const int rl = wm + ti * 16 + crow + i;
        const int cl = wn + tj * 16 + ccol;
        Cs[rl][cl] = (accs[ti][tj][i] - tau_s[rl]) > 0.f ? 1.f : 0.f;
      }
  __syncthreads();
  if (tid < 64) {
    float as = 0.f;
    for (int c = 0; c < BLK_N; ++c) as += Cs[tid][c];
    atomicAdd(&tact[m0 + tid], as);
  }
}

// ---------------------------------------------------------------- activity finalize
__global__ __launch_bounds__(256) void k_finalize_act(const float* __restrict__ twc,
                                                      const float* __restrict__ tact,
                                                      float* __restrict__ outp) {
  __shared__ float r1[256], r2[256];
  const int tid = threadIdx.x;
  const int m = blockIdx.x * 256 + tid;
  float w = twc[m], a = tact[m];
  outp[OFF_ACT + m] = a * (1.f / N_TOT);
  outp[OFF_GMX + m] = a > 0.f ? 1.f : 0.f;
  r1[tid] = w; r2[tid] = a; __syncthreads();
  for (int s = 128; s > 0; s >>= 1) {
    if (tid < s) { r1[tid] += r1[tid + s]; r2[tid] += r2[tid + s]; }
    __syncthreads();
  }
  if (tid == 0) {
    atomicAdd(&outp[OFF_SC + 2], r1[0] * (1.f / M_TOT));
    atomicAdd(&outp[OFF_SC + 3], r2[0] * (1.f / M_TOT));
  }
}

// ---------------------------------------------------------------- out = (t @ wc)/den; B from pre-transposed wcT
__global__ __launch_bounds__(256) void k_out_gemm(const bf16_t* __restrict__ tb,
                                                  const bf16_t* __restrict__ wcT, // [D][N]
                                                  const float* __restrict__ twc,
                                                  float* __restrict__ outp) {
  __shared__ __align__(16) char smraw[SM_OUT];
  __shared__ float den_s[BLK_M];

  const int m0 = blockIdx.x * BLK_M, d0 = blockIdx.y * BLK_ND;
  const int tid = threadIdx.x, wave = tid >> 5, lane = tid & 31;
  const int wm = (wave & 1) * 32, wn = (wave >> 1) * 64;   // wave tile 32x64 -> 8 WMMAs
  const int ar = tid >> 2, ac = (tid & 3) * 8;             // A: 64x32
  const int brr = tid >> 2, bcc = (tid & 3) * 8;           // B: 256x32 in 4 reps
  const bf16_t* ap = tb + (size_t)(m0 + ar) * N_TOT + ac;
  const bf16_t* bp = wcT + (size_t)(d0 + brr) * N_TOT + bcc;
  if (tid < BLK_M) den_s[tid] = fmaxf(sqrtf(twc[m0 + tid] + 1e-6f), 1e-3f);

  { // prologue: buffer 0 (layout: A[2] | B[2])
    bf16_t* A0 = (bf16_t*)smraw;
    bf16_t* B0 = (bf16_t*)(smraw + 2 * A_BYTES);
    v8bf av = *(const v8bf*)(ap);
    *(v8bf*)(A0 + ar * LDA + ac) = av;
#pragma unroll
    for (int rep = 0; rep < 4; ++rep) {
      v8bf bv = *(const v8bf*)(bp + (size_t)rep * 64 * N_TOT);
      *(v8bf*)(B0 + (brr + rep * 64) * LDA + bcc) = bv;
    }
  }
  __syncthreads();

  v8f acc[2][4] = {};
  for (int k0 = 0; k0 < N_TOT; k0 += KSTEP) {
    const int cur = (k0 / KSTEP) & 1, nxt = cur ^ 1;
    const bool more = (k0 + KSTEP) < N_TOT;
    bf16_t* Acur = (bf16_t*)(smraw + cur * A_BYTES);
    bf16_t* Anxt = (bf16_t*)(smraw + nxt * A_BYTES);
    bf16_t* Bcur = (bf16_t*)(smraw + 2 * A_BYTES + cur * BD_BYTES);
    bf16_t* Bnxt = (bf16_t*)(smraw + 2 * A_BYTES + nxt * BD_BYTES);
    v8bf av = {}, bv[4] = {};
    if (more) {
      av = *(const v8bf*)(ap + k0 + KSTEP);
#pragma unroll
      for (int rep = 0; rep < 4; ++rep)
        bv[rep] = *(const v8bf*)(bp + (size_t)rep * 64 * N_TOT + k0 + KSTEP);
    }
    if (k0 + 2 * KSTEP < N_TOT) {  // prefetch 2 tiles ahead
      __builtin_prefetch(ap + k0 + 2 * KSTEP, 0, 1);
      __builtin_prefetch(bp + k0 + 2 * KSTEP, 0, 1);
    }
    v16bf a0 = frag_ld(Acur + (wm + 0)  * LDA, lane);
    v16bf a1 = frag_ld(Acur + (wm + 16) * LDA, lane);
#pragma unroll
    for (int tj = 0; tj < 4; ++tj) {
      v16bf b = frag_ld(Bcur + (wn + tj * 16) * LDA, lane);
      acc[0][tj] = wmma_bf16(a0, b, acc[0][tj]);
      acc[1][tj] = wmma_bf16(a1, b, acc[1][tj]);
    }
    if (more) {
      *(v8bf*)(Anxt + ar * LDA + ac) = av;
#pragma unroll
      for (int rep = 0; rep < 4; ++rep)
        *(v8bf*)(Bnxt + (brr + rep * 64) * LDA + bcc) = bv[rep];
    }
    __syncthreads();
  }

  const int crow = (lane >> 4) * 8, ccol = lane & 15;
#pragma unroll
  for (int ti = 0; ti < 2; ++ti)
#pragma unroll
    for (int tj = 0; tj < 4; ++tj)
#pragma unroll
      for (int i = 0; i < 8; ++i) {
        const int rl = wm + ti * 16 + crow + i;
        const int cl = wn + tj * 16 + ccol;
        float v = acc[ti][tj][i] / den_s[rl];
        outp[OFF_OUT + (size_t)(m0 + rl) * D_DIM + d0 + cl] = (float)(bf16_t)v;
      }
}

// ---------------------------------------------------------------- launcher
extern "C" void kernel_launch(void* const* d_in, const int* in_sizes, int n_in,
                              void* d_out, int out_size, void* d_ws, size_t ws_size,
                              hipStream_t stream) {
  const float* x       = (const float*)d_in[0];
  const float* h       = (const float*)d_in[1];
  const float* emb     = (const float*)d_in[2];
  const float* tau_off = (const float*)d_in[3];
  const float* w_read  = (const float*)d_in[4];
  const float* w_write = (const float*)d_in[5];
  float* outp = (float*)d_out;
  char*  ws   = (char*)d_ws;

  const size_t ND2 = (size_t)N_TOT * D_DIM * 2;   // 64 MB
  const size_t MD2 = (size_t)M_TOT * D_DIM * 2;   // 8 MB
  const size_t MN2 = (size_t)M_TOT * N_TOT * 2;   // 256 MB
  bf16_t* emb_u = (bf16_t*)(ws);
  bf16_t* rcn   = (bf16_t*)(ws + ND2);
  bf16_t* wcn   = (bf16_t*)(ws + 2 * ND2);
  bf16_t* wcT   = (bf16_t*)(ws + 3 * ND2);
  bf16_t* hb    = (bf16_t*)(ws + 4 * ND2);
  bf16_t* xb    = (bf16_t*)(ws + 4 * ND2 + MD2);
  bf16_t* tbuf  = (bf16_t*)(ws + 4 * ND2 + 2 * MD2);
  float*  stats = (float*)(ws + 4 * ND2 + 2 * MD2 + MN2);
  float* s_sum  = stats;
  float* sq_sum = s_sum + M_TOT;
  float* colsum = sq_sum + M_TOT;
  float* twc    = colsum + N_TOT;
  float* tact   = twc + M_TOT;
  float* tau    = tact + M_TOT;
  float* sstd   = tau + M_TOT;

  (void)hipMemsetAsync(stats, 0, (size_t)(4 * M_TOT + N_TOT) * sizeof(float), stream);
  (void)hipMemsetAsync(outp + OFF_SC + 2, 0, 2 * sizeof(float), stream);

  k_rownorm_bf16<<<N_TOT, 256, 0, stream>>>(emb,     emb_u, 1);
  k_rownorm_bf16<<<N_TOT, 256, 0, stream>>>(w_read,  rcn,   1);
  k_rownorm_bf16<<<N_TOT, 256, 0, stream>>>(w_write, wcn,   1);
  k_rownorm_bf16<<<M_TOT, 256, 0, stream>>>(h,       hb,    0);
  k_rownorm_bf16<<<M_TOT, 256, 0, stream>>>(x,       xb,    0);
  k_transpose<<<dim3(N_TOT / 64, D_DIM / 64), 256, 0, stream>>>(wcn, wcT);

  k_stats_gemm<<<dim3(M_TOT / BLK_M, N_TOT / BLK_N), 256, 0, stream>>>(
      hb, emb_u, s_sum, sq_sum, colsum);
  k_finalize_stats<<<1, 256, 0, stream>>>(s_sum, sq_sum, colsum, tau_off, tau, sstd, outp);

  k_gate_gemm<<<dim3(M_TOT / BLK_M, N_TOT / BLK_N), 256, 0, stream>>>(
      hb, xb, emb_u, rcn, tau, sstd, tbuf, twc, tact);
  k_finalize_act<<<M_TOT / 256, 256, 0, stream>>>(twc, tact, outp);

  k_out_gemm<<<dim3(M_TOT / BLK_M, D_DIM / BLK_ND), 256, 0, stream>>>(tbuf, wcT, twc, outp);
}
